// GAT_processor_89601607729412
// MI455X (gfx1250) — matrix operand order repaired and checked by reading, not compiled, not run on hardware
//
#include <hip/hip_runtime.h>
#include <stddef.h>

#define D_ 128
#define EPS_ 1e-5f

typedef __attribute__((ext_vector_type(16))) __bf16 v16bf;
typedef __attribute__((ext_vector_type(8)))  float  v8f;
typedef __attribute__((ext_vector_type(4)))  unsigned int u32x4;
typedef __attribute__((ext_vector_type(4)))  float  f32x4;

union Frag16 {
  v16bf v;
  u32x4 u[2];
  __bf16 e[16];
};

__device__ __forceinline__ v8f wmma_bf16(const Frag16& a, const Frag16& b, v8f c) {
  // D = A(16x32 bf16) * B(32x16 bf16) + C(16x16 f32)
  return __builtin_amdgcn_wmma_f32_16x16x32_bf16(
      /*neg_a=*/false, a.v, /*neg_b=*/false, b.v,
      /*c_mod=*/(short)0, c, /*reuse_a=*/false, /*reuse_b=*/false);
}

// A-fragment: lane (g = lane>>4, m = lane&15) holds row m, K = k0+g*8..+7 (lo)
// and K = k0+16+g*8..+7 (hi) -- per ISA 16-bit A 16x32 layout.
__device__ __forceinline__ void loadFragA(Frag16& f, const __bf16* row, int k0, int g) {
  f.u[0] = *(const u32x4*)(row + k0 + g * 8);
  f.u[1] = *(const u32x4*)(row + k0 + 16 + g * 8);
}
// B-fragment from transposed weight Wt[n][k] (row-major 128x128):
// lane holds column n = (lane&15)+16*nt, K = k0+g*8..+7 (lo) and k0+16+g*8..+7 (hi).
__device__ __forceinline__ void loadFragB(Frag16& f, const __bf16* W, int n, int k0, int g) {
  const __bf16* p = W + (size_t)n * D_ + k0 + g * 8;
  f.u[0] = *(const u32x4*)(p);
  f.u[1] = *(const u32x4*)(p + 16);
}

__device__ __forceinline__ void atomicMaxF(float* addr, float v) {
  if (v >= 0.f) atomicMax((int*)addr, __float_as_int(v));
  else          atomicMin((unsigned int*)addr, __float_as_uint(v));
}

// ---------------------------------------------------------------- weights prep
// Wt[l][mat][n][k] = W_mat[l][k][n] in bf16 (mat 0..2 = A,B,C); Vt[l][n][k] = V[l][k][n]
__global__ void prepWeightsK(const float* __restrict__ A, const float* __restrict__ B,
                             const float* __restrict__ C, const float* __restrict__ V,
                             __bf16* __restrict__ Wt, __bf16* __restrict__ Vt, int L) {
  size_t idx = (size_t)blockIdx.x * blockDim.x + threadIdx.x;
  size_t total = (size_t)L * 4 * D_ * D_;
  if (idx >= total) return;
  int l   = (int)(idx / (4 * D_ * D_));
  int rem = (int)(idx % (4 * D_ * D_));
  int mat = rem / (D_ * D_);
  int q   = rem % (D_ * D_);
  int n = q / D_, k = q % D_;
  const float* s = (mat == 0) ? A : (mat == 1) ? B : (mat == 2) ? C : V;
  float v = s[(size_t)l * D_ * D_ + (size_t)k * D_ + n];
  if (mat < 3) Wt[((size_t)l * 3 + mat) * D_ * D_ + (size_t)n * D_ + k] = (__bf16)v;
  else         Vt[(size_t)l * D_ * D_ + (size_t)n * D_ + k] = (__bf16)v;
}

__global__ void convertK(const float* __restrict__ x, float* __restrict__ xf,
                         __bf16* __restrict__ xb, size_t n) {
  size_t i = (size_t)blockIdx.x * blockDim.x + threadIdx.x;
  if (i < n) { float v = x[i]; xf[i] = v; xb[i] = (__bf16)v; }
}

__global__ void initNodeK(float* __restrict__ maxb, float* __restrict__ denom, int N) {
  int i = blockIdx.x * blockDim.x + threadIdx.x;
  if (i < N) { maxb[i] = -__builtin_inff(); denom[i] = 0.f; }
}

// ---------------------------------------------------------------- edge GEMM
// Per 16-edge tile: e_hat = hs@A + hd@B + e@C  (bf16 WMMA, f32 acc),
// fused logits = leaky_relu(e_hat) . a and segment-max into maxb[dst].
__global__ __launch_bounds__(256) void edgeGemmK(
    const __bf16* __restrict__ hb, const __bf16* __restrict__ eb,
    const int* __restrict__ src, const int* __restrict__ dst,
    const __bf16* __restrict__ WtL, const float* __restrict__ attnL,
    float* __restrict__ ehat, float* __restrict__ logits,
    float* __restrict__ maxb, int E) {
  __shared__ __bf16 sW[3 * D_ * D_];   // 96 KB: At, Bt, Ct for this layer
  {
    const u32x4* g4 = (const u32x4*)WtL;
    u32x4* s4 = (u32x4*)sW;
    for (int i = threadIdx.x; i < 3 * D_ * D_ / 8; i += blockDim.x) s4[i] = g4[i];
  }
  __syncthreads();

  const int wave = threadIdx.x >> 5;
  const int lane = threadIdx.x & 31;
  const int g = lane >> 4, cn = lane & 15;
  const long tile = (long)blockIdx.x * 8 + wave;
  const long m0 = tile * 16;
  if (m0 >= E) return;

  long myRow = m0 + cn; if (myRow >= E) myRow = E - 1;
  const int sIdx = src[myRow];
  const int dIdx = dst[myRow];

  v8f acc[8] = {};
  for (int k0 = 0; k0 < D_; k0 += 32) {
    Frag16 aS, aD, aE;
    loadFragA(aS, hb + (size_t)sIdx * D_, k0, g);
    loadFragA(aD, hb + (size_t)dIdx * D_, k0, g);
    loadFragA(aE, eb + (size_t)myRow * D_, k0, g);
#pragma unroll
    for (int nt = 0; nt < 8; ++nt) {
      Frag16 w;
      loadFragB(w, sW,              nt * 16 + cn, k0, g);
      acc[nt] = wmma_bf16(aS, w, acc[nt]);
      loadFragB(w, sW + D_ * D_,    nt * 16 + cn, k0, g);
      acc[nt] = wmma_bf16(aD, w, acc[nt]);
      loadFragB(w, sW + 2 * D_ * D_, nt * 16 + cn, k0, g);
      acc[nt] = wmma_bf16(aE, w, acc[nt]);
    }
  }

  float av[8];
#pragma unroll
  for (int nt = 0; nt < 8; ++nt) av[nt] = attnL[nt * 16 + cn];

  // C/D layout: lane half g covers rows m0+g*8 .. +7, column nt*16+cn.
#pragma unroll
  for (int r = 0; r < 8; ++r) {
    long row = m0 + g * 8 + r;
    float ps = 0.f;
#pragma unroll
    for (int nt = 0; nt < 8; ++nt) {
      float x = acc[nt][r];
      if (row < E) ehat[row * D_ + nt * 16 + cn] = x;
      float lx = (x > 0.f) ? x : 0.2f * x;
      ps += lx * av[nt];
    }
    // reduce over 16 columns within each half-wave
    for (int off = 1; off < 16; off <<= 1) ps += __shfl_xor(ps, off, 32);
    if (cn == 0 && row < E) {
      logits[row] = ps;
      atomicMaxF(maxb + dst[row], ps);
    }
  }
}

__global__ void expWK(const float* __restrict__ logits, const int* __restrict__ dst,
                      const float* __restrict__ maxb, float* __restrict__ wbuf,
                      float* __restrict__ denom, int E) {
  int e = blockIdx.x * blockDim.x + threadIdx.x;
  if (e < E) {
    int d = dst[e];
    float wv = __expf(logits[e] - maxb[d]);
    wbuf[e] = wv;
    atomicAdd(denom + d, wv);
  }
}

__global__ void alphaK(const float* __restrict__ wbuf, const int* __restrict__ dst,
                       const float* __restrict__ denom, float* __restrict__ alpha, int E) {
  int e = blockIdx.x * blockDim.x + threadIdx.x;
  if (e < E) {
    float dn = denom[dst[e]];
    alpha[e] = wbuf[e] / (dn > 0.f ? dn : 1.f);
  }
}

// s[dst] += alpha * h[src]  (one block per edge, one thread per feature)
__global__ void scatterK(const float* __restrict__ alpha, const int* __restrict__ src,
                         const int* __restrict__ dst, const float* __restrict__ hCur,
                         float* __restrict__ sbuf, int E) {
  int e = blockIdx.x;
  int c = threadIdx.x;
  float al = alpha[e];
  int si = src[e], d = dst[e];
  atomicAdd(&sbuf[(size_t)d * D_ + c], al * hCur[(size_t)si * D_ + c]);
}

// ---------------------------------------------------------------- node GEMM
// hpre = h + s @ V   (s converted f32->bf16 on the fly)
__global__ __launch_bounds__(256) void nodeGemmK(
    const float* __restrict__ sIn, const float* __restrict__ hCur,
    const __bf16* __restrict__ VtL, float* __restrict__ hpre, int N) {
  __shared__ __bf16 sV[D_ * D_];   // 32 KB
  {
    const u32x4* g4 = (const u32x4*)VtL;
    u32x4* s4 = (u32x4*)sV;
    for (int i = threadIdx.x; i < D_ * D_ / 8; i += blockDim.x) s4[i] = g4[i];
  }
  __syncthreads();

  const int wave = threadIdx.x >> 5;
  const int lane = threadIdx.x & 31;
  const int g = lane >> 4, cn = lane & 15;
  const long tile = (long)blockIdx.x * 8 + wave;
  const long m0 = tile * 16;
  if (m0 >= N) return;

  long myRow = m0 + cn; if (myRow >= N) myRow = N - 1;
  const float* rp = sIn + (size_t)myRow * D_;

  v8f acc[8] = {};
  for (int k0 = 0; k0 < D_; k0 += 32) {
    Frag16 a;
    f32x4 l0 = *(const f32x4*)(rp + k0 + g * 8);
    f32x4 l1 = *(const f32x4*)(rp + k0 + g * 8 + 4);
    f32x4 h0 = *(const f32x4*)(rp + k0 + 16 + g * 8);
    f32x4 h1 = *(const f32x4*)(rp + k0 + 16 + g * 8 + 4);
#pragma unroll
    for (int i = 0; i < 4; ++i) {
      a.e[i]      = (__bf16)l0[i];
      a.e[4 + i]  = (__bf16)l1[i];
      a.e[8 + i]  = (__bf16)h0[i];
      a.e[12 + i] = (__bf16)h1[i];
    }
#pragma unroll
    for (int nt = 0; nt < 8; ++nt) {
      Frag16 w;
      loadFragB(w, sV, nt * 16 + cn, k0, g);
      acc[nt] = wmma_bf16(a, w, acc[nt]);
    }
  }

#pragma unroll
  for (int r = 0; r < 8; ++r) {
    long row = m0 + g * 8 + r;
    if (row < N) {
#pragma unroll
      for (int nt = 0; nt < 8; ++nt) {
        size_t i = (size_t)row * D_ + nt * 16 + cn;
        hpre[i] = hCur[i] + acc[nt][r];
      }
    }
  }
}

// ---------------------------------------------------------------- BN helpers
__global__ void colStatsK(const float* __restrict__ x, float* __restrict__ sum,
                          float* __restrict__ sq, int rows) {
  int c = threadIdx.x;
  size_t r0 = (size_t)blockIdx.x * 512;
  size_t rend = r0 + 512; if (rend > (size_t)rows) rend = rows;
  float s = 0.f, q = 0.f;
  for (size_t r = r0; r < rend; ++r) { float v = x[r * D_ + c]; s += v; q += v * v; }
  atomicAdd(sum + c, s);
  atomicAdd(sq + c, q);
}

// acc[i] = base[i] + acc[i] (residual) while accumulating column stats
__global__ void colStatsAddK(float* __restrict__ accb, const float* __restrict__ base,
                             float* __restrict__ sum, float* __restrict__ sq, int rows) {
  int c = threadIdx.x;
  size_t r0 = (size_t)blockIdx.x * 512;
  size_t rend = r0 + 512; if (rend > (size_t)rows) rend = rows;
  float s = 0.f, q = 0.f;
  for (size_t r = r0; r < rend; ++r) {
    size_t i = r * D_ + c;
    float v = base[i] + accb[i];
    accb[i] = v;
    s += v; q += v * v;
  }
  atomicAdd(sum + c, s);
  atomicAdd(sq + c, q);
}

__global__ void bnReluK(const float* __restrict__ pre, const float* __restrict__ sum,
                        const float* __restrict__ sq, const float* __restrict__ gamma,
                        const float* __restrict__ beta, float* __restrict__ outF,
                        __bf16* __restrict__ outB, size_t n, float invRows, int writeB) {
  size_t i = (size_t)blockIdx.x * blockDim.x + threadIdx.x;
  if (i < n) {
    int c = (int)(i & (D_ - 1));
    float mu = sum[c] * invRows;
    float var = sq[c] * invRows - mu * mu;
    float x = (pre[i] - mu) * rsqrtf(var + EPS_) * gamma[c] + beta[c];
    x = (x > 0.f) ? x : 0.f;
    outF[i] = x;
    if (writeB) outB[i] = (__bf16)x;
  }
}

// ---------------------------------------------------------------- launcher
extern "C" void kernel_launch(void* const* d_in, const int* in_sizes, int n_in,
                              void* d_out, int out_size, void* d_ws, size_t ws_size,
                              hipStream_t stream) {
  const float* h_in = (const float*)d_in[0];
  const float* e_in = (const float*)d_in[1];
  const int*   src  = (const int*)d_in[2];
  const int*   dst  = (const int*)d_in[3];
  const float* A    = (const float*)d_in[4];
  const float* B    = (const float*)d_in[5];
  const float* C    = (const float*)d_in[6];
  const float* V    = (const float*)d_in[7];
  const float* attn = (const float*)d_in[8];
  const float* gh   = (const float*)d_in[9];
  const float* bh   = (const float*)d_in[10];
  const float* ge   = (const float*)d_in[11];
  const float* be   = (const float*)d_in[12];

  const int N = in_sizes[0] / D_;
  const int E = in_sizes[2];
  const int L = in_sizes[4] / (D_ * D_);

  char* p = (char*)d_ws;
  auto carve = [&](size_t bytes) -> void* {
    void* r = (void*)p;
    p += (bytes + 255) & ~(size_t)255;
    return r;
  };
  __bf16* Wt    = (__bf16*)carve((size_t)L * 3 * D_ * D_ * sizeof(__bf16));
  __bf16* Vt    = (__bf16*)carve((size_t)L * D_ * D_ * sizeof(__bf16));
  __bf16* hb    = (__bf16*)carve((size_t)N * D_ * sizeof(__bf16));
  __bf16* ebb   = (__bf16*)carve((size_t)E * D_ * sizeof(__bf16));
  float*  h_cur = (float*)carve((size_t)N * D_ * sizeof(float));
  float*  hpre  = (float*)carve((size_t)N * D_ * sizeof(float));
  float*  ebuf  = (float*)carve((size_t)E * D_ * sizeof(float));
  float*  ehat  = (float*)carve((size_t)E * D_ * sizeof(float));
  float*  sbuf  = (float*)carve((size_t)N * D_ * sizeof(float));
  float*  logA  = (float*)carve((size_t)E * sizeof(float));   // logits then alpha
  float*  wbuf  = (float*)carve((size_t)E * sizeof(float));
  float*  maxb  = (float*)carve((size_t)N * sizeof(float));
  float*  denom = (float*)carve((size_t)N * sizeof(float));
  float*  stats = (float*)carve(4 * D_ * sizeof(float)); // hsum,hsq,esum,esq

  float* out_h = (float*)d_out;
  float* out_e = out_h + (size_t)N * D_;

  // One-time (per call) prep: transposed bf16 weights + bf16/f32 working copies.
  {
    size_t tw = (size_t)L * 4 * D_ * D_;
    prepWeightsK<<<(unsigned)((tw + 255) / 256), 256, 0, stream>>>(A, B, C, V, Wt, Vt, L);
    size_t nh = (size_t)N * D_;
    convertK<<<(unsigned)((nh + 255) / 256), 256, 0, stream>>>(h_in, h_cur, hb, nh);
    size_t ne = (size_t)E * D_;
    convertK<<<(unsigned)((ne + 255) / 256), 256, 0, stream>>>(e_in, ebuf, ebb, ne);
  }

  const long tilesE = ((long)E + 15) / 16;
  const long tilesN = ((long)N + 15) / 16;
  const unsigned blkE = (unsigned)((tilesE + 7) / 8);
  const unsigned blkN = (unsigned)((tilesN + 7) / 8);

  for (int l = 0; l < L; ++l) {
    const int last = (l == L - 1);

    hipMemsetAsync(sbuf, 0, (size_t)N * D_ * sizeof(float), stream);
    hipMemsetAsync(stats, 0, 4 * D_ * sizeof(float), stream);
    initNodeK<<<(N + 255) / 256, 256, 0, stream>>>(maxb, denom, N);

    edgeGemmK<<<blkE, 256, 0, stream>>>(hb, ebb, src, dst,
                                        Wt + (size_t)l * 3 * D_ * D_,
                                        attn + (size_t)l * D_,
                                        ehat, logA, maxb, E);

    expWK<<<(E + 255) / 256, 256, 0, stream>>>(logA, dst, maxb, wbuf, denom, E);
    alphaK<<<(E + 255) / 256, 256, 0, stream>>>(wbuf, dst, denom, logA, E);
    scatterK<<<(unsigned)E, D_, 0, stream>>>(logA, src, dst, h_cur, sbuf, E);

    nodeGemmK<<<blkN, 256, 0, stream>>>(sbuf, h_cur, Vt + (size_t)l * D_ * D_, hpre, N);

    colStatsK<<<(unsigned)((N + 511) / 512), D_, 0, stream>>>(hpre, stats, stats + D_, N);
    colStatsAddK<<<(unsigned)((E + 511) / 512), D_, 0, stream>>>(ehat, ebuf,
                                                                 stats + 2 * D_, stats + 3 * D_, E);

    size_t nh = (size_t)N * D_;
    bnReluK<<<(unsigned)((nh + 255) / 256), 256, 0, stream>>>(
        hpre, stats, stats + D_, gh + (size_t)l * D_, bh + (size_t)l * D_,
        last ? out_h : h_cur, hb, nh, 1.f / (float)N, last ? 0 : 1);

    size_t ne = (size_t)E * D_;
    bnReluK<<<(unsigned)((ne + 255) / 256), 256, 0, stream>>>(
        ehat, stats + 2 * D_, stats + 3 * D_, ge + (size_t)l * D_, be + (size_t)l * D_,
        last ? out_e : ebuf, ebb, ne, 1.f / (float)E, last ? 0 : 1);
  }

  (void)n_in; (void)out_size; (void)ws_size;
}
